// LocalSelfAttentionHeadSum_32959579030107
// MI455X (gfx1250) — matrix operand order repaired and checked by reading, not compiled, not run on hardware
//
#include <hip/hip_runtime.h>
#include <hip/hip_bf16.h>

// ---------------------------------------------------------------------------
// CDNA5 (gfx1250) fused local self-attention, bf16 WMMA everywhere.
// Q stored transposed (196x512), K stored transposed+padded (1792x512, pad
// rows zeroed), V stored padded (512x1792, pad cols zeroed) so every WMMA
// fragment load (global or LDS) is a contiguous aligned b128 access.
// ---------------------------------------------------------------------------

typedef __attribute__((ext_vector_type(16))) __bf16 v16bf;
typedef __attribute__((ext_vector_type(8)))  float  v8f;

union Frag16 {
  unsigned short u[16];
  uint4          q[2];
  v16bf          v;
};

union Pack8 {
  unsigned short s[8];
  uint4          v;
};

__device__ __forceinline__ unsigned short f2bf(float f) {
  unsigned int x = __float_as_uint(f);
  x += 0x7FFFu + ((x >> 16) & 1u);
  return (unsigned short)(x >> 16);
}

// ------------------------------ weight convert ------------------------------
__global__ void cvt_f32_to_bf16_k(const float* __restrict__ s,
                                  unsigned short* __restrict__ d, int n) {
  int i = blockIdx.x * blockDim.x + threadIdx.x;
  if (i < n) d[i] = f2bf(s[i]);
}

// ------------------------- generic 128x64 WMMA GEMM -------------------------
// C[M x *] = A[M x K](bf16) * B[K x ldb] (+bias); optional fused second output
// (A2/bias2/C2) sharing the staged B tile so B is read from HBM once.
// b_is_f32 : B is f32, converted to bf16 while staging into LDS.
// c_is_f32 : outputs stored f32, else bf16 bits.
// c_trans  : first output stored transposed, C[col*ldc + row].
// Columns in [Nvalid, Nstore) are zero-filled (padding for the consumer).
__global__ __launch_bounds__(256) void gemm_bf16_128x64(
    const unsigned short* __restrict__ A,
    const unsigned short* __restrict__ A2,       // nullptr -> single output
    const void* __restrict__ Bbase, int b_is_f32, long long strideB_bytes,
    const float* __restrict__ bias, const float* __restrict__ bias2,
    void* __restrict__ Cbase, void* __restrict__ C2base,
    int c_is_f32, int c_trans,
    long long strideC_bytes, long long strideC2_bytes,
    int K, int Nvalid, int Nstore, int ldb, int ldc, int ldc2)
{
  __shared__ __align__(16) unsigned short Asl [128][32];
  __shared__ __align__(16) unsigned short Asl2[128][32];
  __shared__ __align__(16) unsigned short BslX[64][40];   // B^T tile, +8 pad

  const int tid   = threadIdx.x;
  const int wid   = tid >> 5;          // 8 waves (wave32)
  const int lane  = tid & 31;
  const int half  = lane >> 4;
  const int lm    = lane & 15;
  const int waveM = wid >> 1;          // 0..3 -> 32-row slices
  const int waveN = wid & 1;           // 0..1 -> 32-col slices
  const int m0    = blockIdx.x * 128;
  const int n0    = blockIdx.y * 64;

  const char* Bb  = (const char*)Bbase + strideB_bytes * (long long)blockIdx.z;
  char*       Cb  = (char*)Cbase       + strideC_bytes * (long long)blockIdx.z;
  char*       C2b = C2base ? (char*)C2base + strideC2_bytes * (long long)blockIdx.z
                           : (char*)0;

  v8f acc [2][2] = {};
  v8f acc2[2][2] = {};

  const int ar  = tid >> 1;            // A stage: 2 threads/row, 16 elems each
  const int as  = (tid & 1) * 16;
  const int bc  = tid & 63;            // B stage: one column per thread
  const int brs = (tid >> 6) * 8;      //          8-row segment
  const int bcol = n0 + bc;
  const bool binb = bcol < Nvalid;

  for (int k0 = 0; k0 < K; k0 += 32) {
    // stage A (and A2) tile: 128x32 bf16, b128 loads
    {
      const uint4* asrc = reinterpret_cast<const uint4*>(
          A + (size_t)(m0 + ar) * (size_t)K + (size_t)(k0 + as));
      uint4* adst = reinterpret_cast<uint4*>(&Asl[ar][as]);
      adst[0] = asrc[0];
      adst[1] = asrc[1];
      if (k0 + 32 < K) __builtin_prefetch(asrc + 4, 0, 0);   // next K-step row
      if (A2) {
        const uint4* a2src = reinterpret_cast<const uint4*>(
            A2 + (size_t)(m0 + ar) * (size_t)K + (size_t)(k0 + as));
        uint4* a2dst = reinterpret_cast<uint4*>(&Asl2[ar][as]);
        a2dst[0] = a2src[0];
        a2dst[1] = a2src[1];
      }
    }
    // stage B tile transposed: BslX[n][k]; wave-coalesced global reads,
    // one aligned b128 LDS store per thread, per-thread bounds guard.
    {
      Pack8 pk;
      if (b_is_f32) {
        const float* Bf = (const float*)Bb;
#pragma unroll
        for (int j = 0; j < 8; ++j) {
          float v = binb ? Bf[(size_t)(k0 + brs + j) * ldb + bcol] : 0.0f;
          pk.s[j] = f2bf(v);
        }
        if (binb && k0 + 32 < K)
          __builtin_prefetch(Bf + (size_t)(k0 + 32 + brs) * ldb + bcol, 0, 0);
      } else {
        const unsigned short* Bh = (const unsigned short*)Bb;
#pragma unroll
        for (int j = 0; j < 8; ++j)
          pk.s[j] = binb ? Bh[(size_t)(k0 + brs + j) * ldb + bcol]
                         : (unsigned short)0;
      }
      *reinterpret_cast<uint4*>(&BslX[bc][brs]) = pk.v;
    }
    __syncthreads();

    // fragments per ISA 7.12.2 layouts — all contiguous b128 LDS reads
    Frag16 fb[2];
#pragma unroll
    for (int nt = 0; nt < 2; ++nt) {
      const unsigned short* brow = &BslX[waveN * 32 + nt * 16 + lm][0];
      fb[nt].q[0] = *reinterpret_cast<const uint4*>(brow + half * 8);
      fb[nt].q[1] = *reinterpret_cast<const uint4*>(brow + 16 + half * 8);
    }
    Frag16 fa[2];
#pragma unroll
    for (int mt = 0; mt < 2; ++mt) {
      const unsigned short* arow = &Asl[waveM * 32 + mt * 16 + lm][0];
      fa[mt].q[0] = *reinterpret_cast<const uint4*>(arow + half * 8);
      fa[mt].q[1] = *reinterpret_cast<const uint4*>(arow + 16 + half * 8);
    }
#pragma unroll
    for (int mt = 0; mt < 2; ++mt)
#pragma unroll
      for (int nt = 0; nt < 2; ++nt)
        acc[mt][nt] = __builtin_amdgcn_wmma_f32_16x16x32_bf16(
            false, fa[mt].v, false, fb[nt].v, (short)0, acc[mt][nt], false, false);

    if (A2) {
      Frag16 fa2[2];
#pragma unroll
      for (int mt = 0; mt < 2; ++mt) {
        const unsigned short* arow = &Asl2[waveM * 32 + mt * 16 + lm][0];
        fa2[mt].q[0] = *reinterpret_cast<const uint4*>(arow + half * 8);
        fa2[mt].q[1] = *reinterpret_cast<const uint4*>(arow + 16 + half * 8);
      }
#pragma unroll
      for (int mt = 0; mt < 2; ++mt)
#pragma unroll
        for (int nt = 0; nt < 2; ++nt)
          acc2[mt][nt] = __builtin_amdgcn_wmma_f32_16x16x32_bf16(
              false, fa2[mt].v, false, fb[nt].v, (short)0, acc2[mt][nt], false, false);
    }
    __syncthreads();
  }

  // store: D layout -> row = vgpr j + 8*half, col = lane&15 (32-bit indexing:
  // per-batch extents are < 2^31 elements)
  auto store_tile = [&](const v8f& a, int mt, int nt, char* cb,
                        const float* bp, int trans, int ldcx) {
    int col = n0 + waveN * 32 + nt * 16 + lm;
    if (col >= Nstore) return;
    bool inb = col < Nvalid;
    int rbase = m0 + waveM * 32 + mt * 16 + half * 8;
#pragma unroll
    for (int j = 0; j < 8; ++j) {
      int row = rbase + j;
      float v = inb ? a[j] + bp[row] : 0.0f;   // pad cols -> exact zero
      unsigned idx = trans ? (unsigned)col * (unsigned)ldcx + (unsigned)row
                           : (unsigned)row * (unsigned)ldcx + (unsigned)col;
      if (c_is_f32) ((float*)cb)[idx] = v;
      else          ((unsigned short*)cb)[idx] = f2bf(v);
    }
  };
#pragma unroll
  for (int mt = 0; mt < 2; ++mt)
#pragma unroll
    for (int nt = 0; nt < 2; ++nt) {
      store_tile(acc[mt][nt], mt, nt, Cb, bias, c_trans, ldc);
      if (A2) store_tile(acc2[mt][nt], mt, nt, C2b, bias2, 0, ldc2);
    }
}

// ------------------------------ fused attention -----------------------------
#define ATT_THW 1764
#define ATT_TP  1792          // padded to a multiple of 32
#define ATT_HW  196
#define ATT_CI  512

__global__ __launch_bounds__(256) void attn_fused_k(
    const unsigned short* __restrict__ qT,   // (B, 196, 512) bf16 (Q^T)
    const unsigned short* __restrict__ kT,   // (B, 1792, 512) bf16, pad rows 0
    const unsigned short* __restrict__ vp,   // (B, 512, 1792) bf16, pad cols 0
    unsigned short* __restrict__ yb)         // (B, 512, 196) bf16
{
  extern __shared__ char smem[];
  unsigned short* q_s = (unsigned short*)smem;                       // 16x512
  float*          S   = (float*)(smem + 16 * ATT_CI * 2);            // 16x1792 f32
  unsigned short* al  = (unsigned short*)(smem + 16 * ATT_CI * 2
                                          + 16 * ATT_TP * 4);        // 16x1792 bf16
  float*          red = (float*)(smem + 16 * ATT_CI * 2
                                 + 16 * ATT_TP * 4 + 16 * ATT_TP * 2); // 16x16

  const int tid  = threadIdx.x;
  const int wid  = tid >> 5;
  const int lane = tid & 31;
  const int half = lane >> 4;
  const int lm   = lane & 15;
  const int q0   = blockIdx.x * 16;
  const long long b = blockIdx.y;

  const unsigned short* qTb = qT + b * (long long)(ATT_HW * ATT_CI);
  const unsigned short* kTb = kT + b * (long long)(ATT_TP * ATT_CI);
  const unsigned short* vpb = vp + b * (long long)(ATT_CI * ATT_TP);
  unsigned short*       ybb = yb + b * (long long)(ATT_CI * ATT_HW);

  // stage Q^T tile (16 x 512, query-major) via b128 loads; invalid rows -> 0
  for (int idx = tid; idx < 16 * ATT_CI / 8; idx += 256) {
    int m = idx >> 6, ch = idx & 63, col = q0 + m;
    uint4 val = make_uint4(0u, 0u, 0u, 0u);
    if (col < ATT_HW)
      val = *reinterpret_cast<const uint4*>(qTb + col * ATT_CI + ch * 8);
    *reinterpret_cast<uint4*>(&q_s[m * ATT_CI + ch * 8]) = val;
  }
  __syncthreads();

  // ---- phase 1: S[q][t] = sum_e q[e][q] * k[e][t] via WMMA ----
  for (int tt = wid; tt < ATT_TP / 16; tt += 8) {
    int  t0   = tt * 16;
    int  tcol = t0 + lm;
    bool tin  = tcol < ATT_THW;
    v8f sacc = {};
    const unsigned short* krow = kTb + tcol * ATT_CI;  // K^T row (pad rows 0)
    const unsigned short* qrow = &q_s[lm * ATT_CI];
    for (int e0 = 0; e0 < ATT_CI; e0 += 32) {
      Frag16 fa, fb;
      fa.q[0] = *reinterpret_cast<const uint4*>(qrow + e0 + half * 8);
      fa.q[1] = *reinterpret_cast<const uint4*>(qrow + e0 + 16 + half * 8);
      fb.q[0] = *reinterpret_cast<const uint4*>(krow + e0 + half * 8);
      fb.q[1] = *reinterpret_cast<const uint4*>(krow + e0 + 16 + half * 8);
      sacc = __builtin_amdgcn_wmma_f32_16x16x32_bf16(
          false, fa.v, false, fb.v, (short)0, sacc, false, false);
    }
#pragma unroll
    for (int j = 0; j < 8; ++j) {
      int row = half * 8 + j;
      S[row * ATT_TP + tcol] = tin ? sacc[j] : -1e30f;   // pad cols -> -inf
    }
  }
  __syncthreads();

  // ---- phase 2: softmax over t (16 threads per query row) ----
  {
    int r = tid >> 4, l16 = tid & 15;
    float mx = -1e30f;
    for (int c = l16; c < ATT_TP; c += 16) mx = fmaxf(mx, S[r * ATT_TP + c]);
    red[r * 16 + l16] = mx;
    __syncthreads();
    mx = red[r * 16];
#pragma unroll
    for (int i = 1; i < 16; ++i) mx = fmaxf(mx, red[r * 16 + i]);
    __syncthreads();
    float sum = 0.f;
    for (int c = l16; c < ATT_TP; c += 16) {
      float e = __expf(S[r * ATT_TP + c] - mx);
      S[r * ATT_TP + c] = e;
      sum += e;
    }
    red[r * 16 + l16] = sum;
    __syncthreads();
    sum = 0.f;
#pragma unroll
    for (int i = 0; i < 16; ++i) sum += red[r * 16 + i];
    float inv = 1.0f / sum;
    for (int c = l16; c < ATT_TP; c += 16)
      al[r * ATT_TP + c] = f2bf(S[r * ATT_TP + c] * inv);  // pad cols -> 0
  }
  __syncthreads();

  // ---- phase 3: Y[e][q] = sum_t v[e][t] * alpha[q][t] via WMMA ----
  {
    int ebase = wid * 64;                  // each wave: 64 e-rows x 16 queries
    v8f yacc[4] = {};
    const unsigned short* arow = &al[lm * ATT_TP];
    for (int t0 = 0; t0 < ATT_TP; t0 += 32) {
      Frag16 fb;                            // B[t][q] = alpha[q][t]
      fb.q[0] = *reinterpret_cast<const uint4*>(arow + t0 + half * 8);
      fb.q[1] = *reinterpret_cast<const uint4*>(arow + t0 + 16 + half * 8);
#pragma unroll
      for (int mt = 0; mt < 4; ++mt) {
        Frag16 fa;                          // A = V rows, contiguous t (pad 0)
        const unsigned short* vrow = vpb + (ebase + mt * 16 + lm) * ATT_TP;
        fa.q[0] = *reinterpret_cast<const uint4*>(vrow + t0 + half * 8);
        fa.q[1] = *reinterpret_cast<const uint4*>(vrow + t0 + 16 + half * 8);
        yacc[mt] = __builtin_amdgcn_wmma_f32_16x16x32_bf16(
            false, fa.v, false, fb.v, (short)0, yacc[mt], false, false);
      }
    }
    int col = q0 + lm;
    if (col < ATT_HW) {
#pragma unroll
      for (int mt = 0; mt < 4; ++mt)
#pragma unroll
        for (int j = 0; j < 8; ++j) {
          int row = ebase + mt * 16 + half * 8 + j;
          ybb[row * ATT_HW + col] = f2bf(yacc[mt][j]);
        }
    }
  }
}

// --------------------------------- launcher ---------------------------------
extern "C" void kernel_launch(void* const* d_in, const int* in_sizes, int n_in,
                              void* d_out, int out_size, void* d_ws, size_t ws_size,
                              hipStream_t stream) {
  const float* x  = (const float*)d_in[0];   // (32,1024,9,14,14)
  const float* Wq = (const float*)d_in[1];   // (512,1024)
  const float* bq = (const float*)d_in[2];
  const float* Wk = (const float*)d_in[3];
  const float* bk = (const float*)d_in[4];
  const float* Wv = (const float*)d_in[5];
  const float* bv = (const float*)d_in[6];
  const float* Wo = (const float*)d_in[7];   // (1024,512)
  const float* bo = (const float*)d_in[8];

  const int B = 32, C = 1024, Ci = 512, HW = 196, THW = 1764, TP = 1792;
  (void)in_sizes; (void)n_in; (void)out_size; (void)ws_size;

  // workspace layout (bf16 elements)
  unsigned short* ws   = (unsigned short*)d_ws;
  unsigned short* wq_b = ws;
  unsigned short* wk_b = wq_b + (size_t)Ci * C;
  unsigned short* wv_b = wk_b + (size_t)Ci * C;
  unsigned short* wo_b = wv_b + (size_t)Ci * C;
  unsigned short* q_b  = wo_b + (size_t)C * Ci;          // Q^T: B x 196 x 512
  unsigned short* k_b  = q_b + (size_t)B * HW * Ci;      // K^T: B x 1792 x 512
  unsigned short* v_b  = k_b + (size_t)B * TP * Ci;      // V  : B x 512 x 1792
  unsigned short* y_b  = v_b + (size_t)B * Ci * TP;      // Y  : B x 512 x 196

  const int nW = Ci * C;
  cvt_f32_to_bf16_k<<<(nW + 255) / 256, 256, 0, stream>>>(Wq, wq_b, nW);
  cvt_f32_to_bf16_k<<<(nW + 255) / 256, 256, 0, stream>>>(Wk, wk_b, nW);
  cvt_f32_to_bf16_k<<<(nW + 255) / 256, 256, 0, stream>>>(Wv, wv_b, nW);
  cvt_f32_to_bf16_k<<<(nW + 255) / 256, 256, 0, stream>>>(Wo, wo_b, nW);

  const long long xStrideB = (long long)C * THW * 4;     // bytes per batch of x

  // Q^T = (Wq * X_item)^T : transposed store, ldc = 512
  gemm_bf16_128x64<<<dim3(Ci / 128, (HW + 63) / 64, B), 256, 0, stream>>>(
      wq_b, nullptr, x + 4 * HW, 1, xStrideB, bq, nullptr,
      q_b, nullptr, /*c_is_f32*/0, /*c_trans*/1,
      (long long)HW * Ci * 2, 0,
      C, HW, /*Nstore*/HW, THW, /*ldc*/Ci, 0);

  // K^T (transposed, padded to 1792 rows) and V (padded ld=1792) fused:
  // one pass over X feeds both WMMA chains, pad region zero-filled.
  gemm_bf16_128x64<<<dim3(Ci / 128, TP / 64, B), 256, 0, stream>>>(
      wk_b, wv_b, x, 1, xStrideB, bk, bv,
      k_b, v_b, /*c_is_f32*/0, /*c_trans*/1,
      (long long)TP * Ci * 2, (long long)Ci * TP * 2,
      C, THW, /*Nstore*/TP, THW, /*ldc*/Ci, /*ldc2*/TP);

  // fused attention: (13 query tiles) x (32 batches), ~185 KB dynamic LDS
  const size_t shmem = (size_t)16 * Ci * 2 + 16 * TP * 4 + 16 * TP * 2 + 16 * 16 * 4;
  attn_fused_k<<<dim3((HW + 15) / 16, B), 256, shmem, stream>>>(q_b, k_b, v_b, y_b);

  // Out = Wo * Y + bo  (f32 output)
  gemm_bf16_128x64<<<dim3(C / 128, (HW + 63) / 64, B), 256, 0, stream>>>(
      wo_b, nullptr, y_b, 0, (long long)Ci * HW * 2, bo, nullptr,
      d_out, nullptr, /*c_is_f32*/1, /*c_trans*/0,
      (long long)C * HW * 4, 0,
      Ci, HW, /*Nstore*/HW, HW, /*ldc*/HW, 0);
}
